// BASELINE_39573828666137
// MI455X (gfx1250) — matrix operand order repaired and checked by reading, not compiled, not run on hardware
//
#include <hip/hip_runtime.h>

// ---------------------------------------------------------------------------
// DeePMD/NequIP-style GNN forward for MI455X (gfx1250, wave32, WMMA).
// Dense layers (mlp3 / resnet3 / embeddings) use V_WMMA_F32_16X16X32_F16
// with f32 accumulation. Weights are pre-swizzled into B-fragment order so
// each lane loads its whole fragment as one contiguous 32B (v16h) load;
// activations are loaded as float4 (global_load_b128) and packed to f16.
// ---------------------------------------------------------------------------

#define RC_    5.0f
#define EPSV   1e-5f
#define CRES   0.89442719f
#define CNEW   0.4472f
#define BPREF  0.632455532f   // sqrt(2/RC)

typedef __attribute__((ext_vector_type(16))) _Float16 v16h;
typedef __attribute__((ext_vector_type(8)))  float    v8f;
typedef __attribute__((ext_vector_type(4)))  float    v4f;
typedef _Float16 h16;

// ---------------- weight packing: f32 [K,N] -> f16 fragment order ----------
// Output element index: ((kb_tile * (N/16) + ntile) * 32 + lane) * 16 + e
// where lane&15 = column-in-tile, k = 16*(e>>3) + 8*(lane>>4) + (e&7) within the 32-K tile.
__global__ void dpa_pack_frag(const float* __restrict__ W, h16* __restrict__ Wp,
                              int K, int N, int Kp) {
    int t = blockIdx.x * blockDim.x + threadIdx.x;
    int total = Kp * N;
    if (t >= total) return;
    int e    = t & 15;
    int lane = (t >> 4) & 31;
    int ft   = t >> 9;                 // kb_tile * NT + ntile
    int NT   = N >> 4;
    int kbt  = ft / NT, nt = ft % NT;
    int k = kbt * 32 + (e >> 3) * 16 + (lane >> 4) * 8 + (e & 7);
    int n = nt * 16 + (lane & 15);
    Wp[t] = (k < K) ? (h16)W[k * N + n] : (h16)0.0f;
}

// ---------------- edge featurization ---------------------------------------
__global__ void dpa_feat(const int* __restrict__ at, const int* __restrict__ idx,
                         const float* __restrict__ len, const float* __restrict__ bw,
                         float* __restrict__ feat, float* __restrict__ ud, long E) {
    long e = (long)blockIdx.x * blockDim.x + threadIdx.x;
    if (e >= E) return;
    int a0 = at[idx[e]], a1 = at[idx[E + e]];
    float x  = len[e];
    float xr = x * (1.0f / RC_);
    float xr2 = xr * xr, xr3 = xr2 * xr, xr6 = xr3 * xr3, xr7 = xr6 * xr, xr8 = xr7 * xr;
    float u = 1.0f - 28.0f * xr6 + 48.0f * xr7 - 21.0f * xr8;
    u = (xr < 1.0f) ? u : 0.0f;
    ud[e] = u;
    float* f = feat + e * 32;
#pragma unroll
    for (int c = 0; c < 4; ++c) {
        f[c]     = (a0 == c) ? 1.0f : 0.0f;
        f[4 + c] = (a1 == c) ? 1.0f : 0.0f;
    }
    float inv = 1.0f / x;
#pragma unroll
    for (int j = 0; j < 16; ++j)
        f[8 + j] = u * BPREF * sinf(bw[j] * x * (1.0f / RC_)) * inv;
#pragma unroll
    for (int j = 24; j < 32; ++j) f[j] = 0.0f;
}

// ---------------- WMMA GEMM: Y = act(X @ W + b [+ Res]) --------------------
// X: [M, K] fp32 row-major (K % 32 == 0, zero padded). Wp: fragment-ordered.
// One wave computes a 16-row strip x all N columns (NT = N/16 tiles).
template <int NT, bool TANH_OUT, bool TANH_IN, bool HAS_RES>
__global__ void dpa_gemm(const float* __restrict__ X, const h16* __restrict__ Wp,
                         const float* __restrict__ bias, const float* __restrict__ Res,
                         float* __restrict__ Y, long M, int K) {
    const int Nn  = NT * 16;
    int wave = threadIdx.x >> 5;
    int lane = threadIdx.x & 31;
    long tile = (long)blockIdx.x * (blockDim.x >> 5) + wave;
    long m0 = tile * 16;
    if (m0 >= M) return;                        // wave-uniform: EXEC stays full
    int r  = lane & 15;
    int kh = lane >> 4;
    long row = m0 + r; if (row >= M) row = M - 1;
    const float* xrow = X + row * (long)K;
    const v16h* wfrag = (const v16h*)Wp;

    v8f acc[NT];
    v8f zero = {0.f, 0.f, 0.f, 0.f, 0.f, 0.f, 0.f, 0.f};
#pragma unroll
    for (int t = 0; t < NT; ++t) acc[t] = zero;

    for (int kb = 0; kb < K; kb += 32) {
        // A fragment: two contiguous 8-float runs (16B-aligned) -> 4x b128
        const v4f* xlo = (const v4f*)(xrow + kb + (kh << 3));
        const v4f* xhi = (const v4f*)(xrow + kb + 16 + (kh << 3));
        v4f x0 = xlo[0], x1 = xlo[1], x2 = xhi[0], x3 = xhi[1];
        v16h a;
#pragma unroll
        for (int i = 0; i < 4; ++i) {
            float f0 = x0[i], f1 = x1[i], f2 = x2[i], f3 = x3[i];
            if (TANH_IN) { f0 = tanhf(f0); f1 = tanhf(f1); f2 = tanhf(f2); f3 = tanhf(f3); }
            a[i]      = (h16)f0;
            a[4 + i]  = (h16)f1;
            a[8 + i]  = (h16)f2;
            a[12 + i] = (h16)f3;
        }
        long fbase = ((long)(kb >> 5) * NT) * 32 + lane;
#pragma unroll
        for (int t = 0; t < NT; ++t) {
            v16h b = wfrag[fbase + t * 32];     // one contiguous 32B load
            acc[t] = __builtin_amdgcn_wmma_f32_16x16x32_f16(
                false, a, false, b, (short)0, acc[t], false, false);
        }
    }
    // C/D layout: VGPR rr, lanes 0-15 -> M=rr, lanes 16-31 -> M=8+rr; N=lane&15
    int  ncl   = lane & 15;
    long mbase = m0 + kh * 8;
#pragma unroll
    for (int t = 0; t < NT; ++t) {
        int ncol = t * 16 + ncl;
        float bv = bias[ncol];
#pragma unroll
        for (int rr = 0; rr < 8; ++rr) {
            long mrow = mbase + rr;
            if (mrow < M) {
                float val = acc[t][rr] + bv;
                if (HAS_RES)  val += Res[mrow * (long)Nn + ncol];
                if (TANH_OUT) val = tanhf(val);
                Y[mrow * (long)Nn + ncol] = val;
            }
        }
    }
}

// ---------------- row layernorm (64 cols) * ud ------------------------------
__global__ void dpa_ln64_scale(const float* __restrict__ X, const float* __restrict__ g,
                               const float* __restrict__ b, const float* __restrict__ ud,
                               float* __restrict__ Y, long M) {
    long r = (long)blockIdx.x * blockDim.x + threadIdx.x;
    if (r >= M) return;
    const float* x = X + r * 64;
    float m = 0.f;
    for (int j = 0; j < 64; ++j) m += x[j];
    m *= (1.0f / 64.0f);
    float var = 0.f;
    for (int j = 0; j < 64; ++j) { float d = x[j] - m; var += d * d; }
    var *= (1.0f / 64.0f);
    float rs = rsqrtf(var + EPSV);
    float u = ud[r];
    for (int j = 0; j < 64; ++j)
        Y[r * 64 + j] = ((x[j] - m) * rs * g[j] + b[j]) * u;
}

// ---------------- radial residual update -----------------------------------
__global__ void dpa_radup(float* __restrict__ rad, const float* __restrict__ X,
                          const float* __restrict__ g, const float* __restrict__ b,
                          const float* __restrict__ ud, long M) {
    long r = (long)blockIdx.x * blockDim.x + threadIdx.x;
    if (r >= M) return;
    const float* x = X + r * 64;
    float m = 0.f;
    for (int j = 0; j < 64; ++j) m += x[j];
    m *= (1.0f / 64.0f);
    float var = 0.f;
    for (int j = 0; j < 64; ++j) { float d = x[j] - m; var += d * d; }
    var *= (1.0f / 64.0f);
    float rs = rsqrtf(var + EPSV);
    float u = ud[r];
    for (int j = 0; j < 64; ++j) {
        float ln = (x[j] - m) * rs * g[j] + b[j];
        rad[r * 64 + j] = CRES * rad[r * 64 + j] + CNEW * u * ln;
    }
}

// ---------------- SE(2) message scatter ------------------------------------
__global__ void dpa_scatter_msg(const float* __restrict__ sq, const float* __restrict__ ud,
                                const float* __restrict__ vec, const int* __restrict__ idx,
                                float* __restrict__ agg, float* __restrict__ cnt, long E) {
    long e = (long)blockIdx.x * blockDim.x + threadIdx.x;
    if (e >= E) return;
    int n = idx[E + e];
    float u  = ud[e];
    float uu = u * u;
    float v0 = vec[e * 3], v1 = vec[e * 3 + 1], v2 = vec[e * 3 + 2];
    float* ag = agg + (long)n * 48;
#pragma unroll
    for (int s = 0; s < 16; ++s) {
        float f = sq[e * 16 + s] * uu;
        atomicAdd(&ag[s * 3 + 0], f * v0);
        atomicAdd(&ag[s * 3 + 1], f * v1);
        atomicAdd(&ag[s * 3 + 2], f * v2);
    }
    atomicAdd(&cnt[n], 1.0f);
}

// ---------------- gram + layernorm -> node_emb ------------------------------
__global__ void dpa_gram(const float* __restrict__ agg, const float* __restrict__ cnt,
                         const float* __restrict__ g, const float* __restrict__ b,
                         float* __restrict__ node, long N) {
    long n = (long)blockIdx.x * blockDim.x + threadIdx.x;
    if (n >= N) return;
    float c = fmaxf(cnt[n], 1.0f);
    float ci = 1.0f / c;
    float a[48];
    for (int i = 0; i < 48; ++i) a[i] = agg[n * 48 + i] * ci;
    float o[128];
    for (int d = 0; d < 16; ++d)
        for (int e8 = 0; e8 < 8; ++e8)
            o[d * 8 + e8] = a[d*3]*a[e8*3] + a[d*3+1]*a[e8*3+1] + a[d*3+2]*a[e8*3+2];
    float m = 0.f;
    for (int j = 0; j < 128; ++j) m += o[j];
    m *= (1.0f / 128.0f);
    float var = 0.f;
    for (int j = 0; j < 128; ++j) { float d = o[j] - m; var += d * d; }
    var *= (1.0f / 128.0f);
    float rs = rsqrtf(var + EPSV);
    for (int j = 0; j < 128; ++j)
        node[n * 128 + j] = (o[j] - m) * rs * g[j] + b[j];
}

// ---------------- hidden init: gem * 0.5*(node[i1]+node[i0]) ----------------
__global__ void dpa_hid_init(const float* __restrict__ gem, const float* __restrict__ node,
                             const int* __restrict__ idx, float* __restrict__ hid, long M) {
    long t = (long)blockIdx.x * blockDim.x + threadIdx.x;
    long tot = M * 128;
    if (t >= tot) return;
    long e = t >> 7;
    int  j = (int)(t & 127);
    int i0 = idx[e], i1 = idx[M + e];
    hid[t] = gem[t] * 0.5f * (node[(long)i1 * 128 + j] + node[(long)i0 * 128 + j]);
}

// ---------------- weighted node scatter (layer) -----------------------------
__global__ void dpa_scatter_wn(const float* __restrict__ ew, const float* __restrict__ node,
                               const int* __restrict__ idx, float* __restrict__ agg2, long E) {
    long e = (long)blockIdx.x * blockDim.x + threadIdx.x;
    if (e >= E) return;
    int n1 = idx[E + e];
    const float* nr = node + (long)n1 * 128;
    float* ag = agg2 + (long)n1 * 128;
    for (int j = 0; j < 128; ++j)
        atomicAdd(&ag[j], ew[e * 128 + j] * nr[j]);
}

// ---------------- node residual update --------------------------------------
__global__ void dpa_node_update(float* __restrict__ node, const float* __restrict__ agg2,
                                const float* __restrict__ cnt, const float* __restrict__ g,
                                const float* __restrict__ b, long N) {
    long n = (long)blockIdx.x * blockDim.x + threadIdx.x;
    if (n >= N) return;
    float c = fmaxf(cnt[n], 1.0f);
    float ci = 1.0f / c;
    float v[128];
    float m = 0.f;
    for (int j = 0; j < 128; ++j) { v[j] = agg2[n * 128 + j] * ci; m += v[j]; }
    m *= (1.0f / 128.0f);
    float var = 0.f;
    for (int j = 0; j < 128; ++j) { float d = v[j] - m; var += d * d; }
    var *= (1.0f / 128.0f);
    float rs = rsqrtf(var + EPSV);
    for (int j = 0; j < 128; ++j) {
        float ln = (v[j] - m) * rs * g[j] + b[j];
        node[n * 128 + j] = CRES * node[n * 128 + j] + CNEW * ln;
    }
}

// ---------------- concat [node[i0] | hid] -> [M,256] ------------------------
__global__ void dpa_cat_nh(const float* __restrict__ node, const int* __restrict__ idx,
                           const float* __restrict__ hid, float* __restrict__ out, long M) {
    long t = (long)blockIdx.x * blockDim.x + threadIdx.x;
    long tot = M * 256;
    if (t >= tot) return;
    long e = t >> 8;
    int  j = (int)(t & 255);
    out[t] = (j < 128) ? node[(long)idx[e] * 128 + j] : hid[e * 128 + (j - 128)];
}

// ---------------- concat [rad | hid] -> [M,192] -----------------------------
__global__ void dpa_cat_rh(const float* __restrict__ rad, const float* __restrict__ hid,
                           float* __restrict__ out, long M) {
    long t = (long)blockIdx.x * blockDim.x + threadIdx.x;
    long tot = M * 192;
    if (t >= tot) return;
    long e = t / 192;
    int  j = (int)(t - e * 192);
    out[t] = (j < 64) ? rad[e * 64 + j] : hid[e * 128 + (j - 64)];
}

// ===========================================================================
extern "C" void kernel_launch(void* const* d_in, const int* in_sizes, int n_in,
                              void* d_out, int out_size, void* d_ws, size_t ws_size,
                              hipStream_t stream) {
    const int*   atom_types = (const int*)  d_in[0];
    const float* env_vec    = (const float*)d_in[1];
    const float* env_len    = (const float*)d_in[2];
    const float* edge_len   = (const float*)d_in[3];
    const int*   env_idx    = (const int*)  d_in[4];
    const int*   edge_idx   = (const int*)  d_in[5];
    const float* bessel_w   = (const float*)d_in[6];
    const float* sr_W1=(const float*)d_in[7],  *sr_b1=(const float*)d_in[8];
    const float* sr_W2=(const float*)d_in[9],  *sr_b2=(const float*)d_in[10];
    const float* sr_W3=(const float*)d_in[11], *sr_b3=(const float*)d_in[12];
    const float* rd_W1=(const float*)d_in[13], *rd_b1=(const float*)d_in[14];
    const float* rd_W2=(const float*)d_in[15], *rd_b2=(const float*)d_in[16];
    const float* rd_W3=(const float*)d_in[17], *rd_b3=(const float*)d_in[18];
    const float* ne_emb_W=(const float*)d_in[19], *ne_emb_b=(const float*)d_in[20];
    const float* ne_nln_g=(const float*)d_in[21], *ne_nln_b=(const float*)d_in[22];
    const float* ne_eln_g=(const float*)d_in[23], *ne_eln_b=(const float*)d_in[24];
    const float* ly_emb_W=(const float*)d_in[25], *ly_emb_b=(const float*)d_in[26];
    const float* ly_hid_W1=(const float*)d_in[27], *ly_hid_b1=(const float*)d_in[28];
    const float* ly_hid_W2=(const float*)d_in[29], *ly_hid_b2=(const float*)d_in[30];
    const float* ly_hid_W3=(const float*)d_in[31], *ly_hid_b3=(const float*)d_in[32];
    const float* ly_rdl_W1=(const float*)d_in[33], *ly_rdl_b1=(const float*)d_in[34];
    const float* ly_rdl_W2=(const float*)d_in[35], *ly_rdl_b2=(const float*)d_in[36];
    const float* ly_rdl_W3=(const float*)d_in[37], *ly_rdl_b3=(const float*)d_in[38];
    const float* ly_nln_g=(const float*)d_in[39], *ly_nln_b=(const float*)d_in[40];
    const float* ly_eln_g=(const float*)d_in[41], *ly_eln_b=(const float*)d_in[42];

    const long N  = in_sizes[0];   // 40000 atoms
    const long E  = in_sizes[2];   // 320000 env edges
    const long E2 = in_sizes[3];   // 160000 edges

    // ---------------- outputs (concatenated in return order) ----------------
    float* out  = (float*)d_out;
    float* envR = out;                       // [E, 64]
    float* edgR = envR + E * 64;             // [E2, 64]
    float* node = edgR + E2 * 64;            // [N, 128]
    float* envH = node + N * 128;            // [E, 128]
    float* edgH = envH + E * 128;            // [E2, 128]

    // ---------------- workspace layout (floats) -----------------------------
    float* ws = (float*)d_ws;
    float* S0    = ws;                 // [E, 256] concat buffer
    float* S1    = S0  + E * 256;      // [E, 128]
    float* S2    = S1  + E * 128;      // [E, 128]
    float* S3    = S2  + E * 128;      // [E, 192]
    float* S4    = S3  + E * 192;      // [E, 64]
    float* S5    = S4  + E * 64;       // [E, 64]
    float* featE = S5  + E * 64;       // [E, 32]
    float* featD = featE + E * 32;     // [E2, 32]
    float* udE   = featD + E2 * 32;    // [E]
    float* udD   = udE + E;            // [E2]
    float* sqE   = udD + E2;           // [E, 16]
    float* agg   = sqE + E * 16;       // [N, 48]
    float* cnt   = agg + N * 48;       // [N]
    float* agg2  = cnt + N;            // [N, 128]
    h16*   hp    = (h16*)(agg2 + N * 128);

    h16* srW1p = hp; hp += 32 * 64;
    h16* srW2p = hp; hp += 64 * 64;
    h16* srW3p = hp; hp += 64 * 16;
    h16* rdW1p = hp; hp += 32 * 64;
    h16* rdW2p = hp; hp += 64 * 64;
    h16* rdW3p = hp; hp += 64 * 64;
    h16* neEp  = hp; hp += 64 * 128;
    h16* lyEp[2];  lyEp[0]  = hp; hp += 64 * 128;  lyEp[1]  = hp; hp += 64 * 128;
    h16* hW1p[2];  hW1p[0]  = hp; hp += 256 * 128; hW1p[1]  = hp; hp += 256 * 128;
    h16* hW2p[2];  hW2p[0]  = hp; hp += 128 * 128; hW2p[1]  = hp; hp += 128 * 128;
    h16* hW3p[2];  hW3p[0]  = hp; hp += 128 * 128; hW3p[1]  = hp; hp += 128 * 128;
    h16* rW1p[2];  rW1p[0]  = hp; hp += 192 * 64;  rW1p[1]  = hp; hp += 192 * 64;
    h16* rW2p[2];  rW2p[0]  = hp; hp += 64 * 64;   rW2p[1]  = hp; hp += 64 * 64;
    h16* rW3p[2];  rW3p[0]  = hp; hp += 64 * 64;   rW3p[1]  = hp; hp += 64 * 64;

#define PACK(Wsrc, Wdst, K, Nc, Kp) \
    dpa_pack_frag<<<((Kp)*(Nc)+255)/256, 256, 0, stream>>>(Wsrc, Wdst, K, Nc, Kp)
#define GGRID(M) ((unsigned)((((M) + 15) / 16 + 3) / 4))
#define EWGRID(T) ((unsigned)(((T) + 255) / 256))

    // ---------------- pack all weights to f16 fragment order ----------------
    PACK(sr_W1, srW1p, 24, 64, 32);   PACK(sr_W2, srW2p, 64, 64, 64);
    PACK(sr_W3, srW3p, 64, 16, 64);
    PACK(rd_W1, rdW1p, 24, 64, 32);   PACK(rd_W2, rdW2p, 64, 64, 64);
    PACK(rd_W3, rdW3p, 64, 64, 64);
    PACK(ne_emb_W, neEp, 64, 128, 64);
    for (int l = 0; l < 2; ++l) {
        PACK(ly_emb_W  + (long)l * 64  * 128, lyEp[l], 64,  128, 64);
        PACK(ly_hid_W1 + (long)l * 256 * 128, hW1p[l], 256, 128, 256);
        PACK(ly_hid_W2 + (long)l * 128 * 128, hW2p[l], 128, 128, 128);
        PACK(ly_hid_W3 + (long)l * 128 * 128, hW3p[l], 128, 128, 128);
        PACK(ly_rdl_W1 + (long)l * 192 * 64,  rW1p[l], 192, 64,  192);
        PACK(ly_rdl_W2 + (long)l * 64  * 64,  rW2p[l], 64,  64,  64);
        PACK(ly_rdl_W3 + (long)l * 64  * 64,  rW3p[l], 64,  64,  64);
    }

    // ---------------- featurization ----------------------------------------
    dpa_feat<<<EWGRID(E),  256, 0, stream>>>(atom_types, env_idx,  env_len,  bessel_w, featE, udE, E);
    dpa_feat<<<EWGRID(E2), 256, 0, stream>>>(atom_types, edge_idx, edge_len, bessel_w, featD, udD, E2);

    // ---------------- env_sqrt = mlp3_sr(feat_env) (raw; *ud fused later) ---
    dpa_gemm<4, true,  false, false><<<GGRID(E), 128, 0, stream>>>(featE, srW1p, sr_b1, nullptr, S4, E, 32);
    dpa_gemm<4, true,  false, false><<<GGRID(E), 128, 0, stream>>>(S4,    srW2p, sr_b2, nullptr, S5, E, 64);
    dpa_gemm<1, false, false, false><<<GGRID(E), 128, 0, stream>>>(S5,    srW3p, sr_b3, nullptr, sqE, E, 64);

    // ---------------- env_radial = ln(mlp3_rd(feat_env)) * ud ---------------
    dpa_gemm<4, true,  false, false><<<GGRID(E), 128, 0, stream>>>(featE, rdW1p, rd_b1, nullptr, S4, E, 32);
    dpa_gemm<4, true,  false, false><<<GGRID(E), 128, 0, stream>>>(S4,    rdW2p, rd_b2, nullptr, S5, E, 64);
    dpa_gemm<4, false, false, false><<<GGRID(E), 128, 0, stream>>>(S5,    rdW3p, rd_b3, nullptr, S4, E, 64);
    dpa_ln64_scale<<<EWGRID(E), 256, 0, stream>>>(S4, ne_eln_g, ne_eln_b, udE, envR, E);

    // ---------------- edge_radial -------------------------------------------
    dpa_gemm<4, true,  false, false><<<GGRID(E2), 128, 0, stream>>>(featD, rdW1p, rd_b1, nullptr, S4, E2, 32);
    dpa_gemm<4, true,  false, false><<<GGRID(E2), 128, 0, stream>>>(S4,    rdW2p, rd_b2, nullptr, S5, E2, 64);
    dpa_gemm<4, false, false, false><<<GGRID(E2), 128, 0, stream>>>(S5,    rdW3p, rd_b3, nullptr, S4, E2, 64);
    dpa_ln64_scale<<<EWGRID(E2), 256, 0, stream>>>(S4, ne_eln_g, ne_eln_b, udD, edgR, E2);

    // ---------------- SE(2) aggregation -> node_emb -------------------------
    hipMemsetAsync(agg, 0, (size_t)N * 48 * sizeof(float), stream);
    hipMemsetAsync(cnt, 0, (size_t)N * sizeof(float), stream);
    dpa_scatter_msg<<<EWGRID(E), 256, 0, stream>>>(sqE, udE, env_vec, env_idx, agg, cnt, E);
    dpa_gram<<<EWGRID(N), 256, 0, stream>>>(agg, cnt, ne_nln_g, ne_nln_b, node, N);

    // ---------------- env_hidden / edge_hidden init -------------------------
    dpa_gemm<8, false, false, false><<<GGRID(E), 128, 0, stream>>>(envR, neEp, ne_emb_b, nullptr, S1, E, 64);
    dpa_hid_init<<<EWGRID(E * 128), 256, 0, stream>>>(S1, node, env_idx, envH, E);
    dpa_gemm<8, false, false, false><<<GGRID(E2), 128, 0, stream>>>(edgR, neEp, ne_emb_b, nullptr, S1, E2, 64);
    dpa_hid_init<<<EWGRID(E2 * 128), 256, 0, stream>>>(S1, node, edge_idx, edgH, E2);

    // ---------------- message-passing layers --------------------------------
    for (int l = 0; l < 2; ++l) {
        // env_weight = env_radial @ ly_emb_W[l] + b ; node update
        dpa_gemm<8, false, false, false><<<GGRID(E), 128, 0, stream>>>(
            envR, lyEp[l], ly_emb_b + (long)l * 128, nullptr, S1, E, 64);
        hipMemsetAsync(agg2, 0, (size_t)N * 128 * sizeof(float), stream);
        dpa_scatter_wn<<<EWGRID(E), 256, 0, stream>>>(S1, node, env_idx, agg2, E);
        dpa_node_update<<<EWGRID(N), 256, 0, stream>>>(
            node, agg2, cnt, ly_nln_g + (long)l * 128, ly_nln_b + (long)l * 128, N);

        // env_hidden = mlp3([node[i0] | env_hidden])
        dpa_cat_nh<<<EWGRID(E * 256), 256, 0, stream>>>(node, env_idx, envH, S0, E);
        dpa_gemm<8, true,  false, false><<<GGRID(E), 128, 0, stream>>>(
            S0, hW1p[l], ly_hid_b1 + (long)l * 128, nullptr, S1, E, 256);
        dpa_gemm<8, true,  false, false><<<GGRID(E), 128, 0, stream>>>(
            S1, hW2p[l], ly_hid_b2 + (long)l * 128, nullptr, S2, E, 128);
        dpa_gemm<8, false, false, false><<<GGRID(E), 128, 0, stream>>>(
            S2, hW3p[l], ly_hid_b3 + (long)l * 128, nullptr, envH, E, 128);

        // edge_hidden = mlp3([node[i0] | edge_hidden])
        dpa_cat_nh<<<EWGRID(E2 * 256), 256, 0, stream>>>(node, edge_idx, edgH, S0, E2);
        dpa_gemm<8, true,  false, false><<<GGRID(E2), 128, 0, stream>>>(
            S0, hW1p[l], ly_hid_b1 + (long)l * 128, nullptr, S1, E2, 256);
        dpa_gemm<8, true,  false, false><<<GGRID(E2), 128, 0, stream>>>(
            S1, hW2p[l], ly_hid_b2 + (long)l * 128, nullptr, S2, E2, 128);
        dpa_gemm<8, false, false, false><<<GGRID(E2), 128, 0, stream>>>(
            S2, hW3p[l], ly_hid_b3 + (long)l * 128, nullptr, edgH, E2, 128);

        // env_radial: resnet3([env_radial | env_hidden]) -> ln -> residual
        dpa_cat_rh<<<EWGRID(E * 192), 256, 0, stream>>>(envR, envH, S3, E);
        dpa_gemm<4, true,  false, false><<<GGRID(E), 128, 0, stream>>>(
            S3, rW1p[l], ly_rdl_b1 + (long)l * 64, nullptr, S4, E, 192);
        dpa_gemm<4, false, false, false><<<GGRID(E), 128, 0, stream>>>(
            S4, rW2p[l], ly_rdl_b2 + (long)l * 64, nullptr, S5, E, 64);
        dpa_gemm<4, false, true,  true ><<<GGRID(E), 128, 0, stream>>>(
            S5, rW3p[l], ly_rdl_b3 + (long)l * 64, S5, S4, E, 64);
        dpa_radup<<<EWGRID(E), 256, 0, stream>>>(
            envR, S4, ly_eln_g + (long)l * 64, ly_eln_b + (long)l * 64, udE, E);

        // edge_radial: same
        dpa_cat_rh<<<EWGRID(E2 * 192), 256, 0, stream>>>(edgR, edgH, S3, E2);
        dpa_gemm<4, true,  false, false><<<GGRID(E2), 128, 0, stream>>>(
            S3, rW1p[l], ly_rdl_b1 + (long)l * 64, nullptr, S4, E2, 192);
        dpa_gemm<4, false, false, false><<<GGRID(E2), 128, 0, stream>>>(
            S4, rW2p[l], ly_rdl_b2 + (long)l * 64, nullptr, S5, E2, 64);
        dpa_gemm<4, false, true,  true ><<<GGRID(E2), 128, 0, stream>>>(
            S5, rW3p[l], ly_rdl_b3 + (long)l * 64, S5, S4, E2, 64);
        dpa_radup<<<EWGRID(E2), 256, 0, stream>>>(
            edgR, S4, ly_eln_g + (long)l * 64, ly_eln_b + (long)l * 64, udD, E2);
    }
#undef PACK
#undef GGRID
#undef EWGRID
}
